// CustomSparseSelfAttention_10307921511056
// MI455X (gfx1250) — matrix operand order repaired and checked by reading, not compiled, not run on hardware
//
#include <hip/hip_runtime.h>

// Problem constants
#define BB   2
#define SS   2048
#define HID  1024
#define HH   16
#define NBQ  32
#define KK   8
#define BLK  64
#define DD   64
#define SLEN (KK * BLK)   // 512 gathered key positions per query block

typedef __attribute__((ext_vector_type(16))) __bf16 v16bf;
typedef __attribute__((ext_vector_type(8)))  float  v8f;
typedef __attribute__((ext_vector_type(4)))  unsigned int u32x4;

union BF16Frag { v16bf v; u32x4 q[2]; };

__device__ __forceinline__ unsigned short f2bf(float f) {
  unsigned int u = __float_as_uint(f);
  u += 0x7FFFu + ((u >> 16) & 1u);   // round-to-nearest-even
  return (unsigned short)(u >> 16);
}

__device__ __forceinline__ v8f wmma_bf16(const BF16Frag& a, const BF16Frag& b, v8f c) {
  // 8 args: (neg_a, A, neg_b, B, c_mod, C, reuse_a, reuse_b)
  return __builtin_amdgcn_wmma_f32_16x16x32_bf16(false, a.v, false, b.v, (short)0, c, false, false);
}

// LDS byte offset of a pointer to __shared__ (flat LDS aperture maps addr[31:0]
// directly to the LDS address, per CDNA5 ISA aperture rules).
__device__ __forceinline__ unsigned lds_off(const void* p) {
  return (unsigned)(unsigned long long)p;
}

// 16B global -> LDS async copy (per active lane), tracked by ASYNCcnt.
__device__ __forceinline__ void async_copy_b128(unsigned lds_dst, const void* gsrc) {
  asm volatile("global_load_async_to_lds_b128 %0, %1, off"
               :: "v"(lds_dst), "v"(gsrc) : "memory");
}

__device__ __forceinline__ void wait_asynccnt0() {
  asm volatile("s_wait_asynccnt 0" ::: "memory");
}

// ---------------------------------------------------------------------------
// Kernel 1: fp32 -> bf16 conversion (grid-stride)
// ---------------------------------------------------------------------------
__global__ void cvt_f32_bf16_kernel(const float* __restrict__ src,
                                    unsigned short* __restrict__ dst, int n) {
  int i = blockIdx.x * blockDim.x + threadIdx.x;
  int stride = gridDim.x * blockDim.x;
  for (; i < n; i += stride) dst[i] = f2bf(src[i]);
}

// ---------------------------------------------------------------------------
// Kernel 2: W (k-major 1024x1024 fp32) -> Wt[n][k] bf16  (tiled transpose)
// ---------------------------------------------------------------------------
__global__ __launch_bounds__(256) void transpose_w_kernel(
    const float* __restrict__ W, unsigned short* __restrict__ Wt) {
  __shared__ unsigned short tile[64][72];
  int bx = blockIdx.x;   // k-tile
  int by = blockIdx.y;   // n-tile
  int tid = threadIdx.x;
  for (int e = tid; e < 4096; e += 256) {
    int r = e >> 6, c = e & 63;     // r: k-local, c: n-local
    tile[r][c] = f2bf(W[(size_t)(bx * 64 + r) * HID + by * 64 + c]);
  }
  __syncthreads();
  for (int e = tid; e < 4096; e += 256) {
    int r = e >> 6, c = e & 63;     // r: n-local, c: k-local
    Wt[(size_t)(by * 64 + r) * HID + bx * 64 + c] = tile[c][r];
  }
}

// ---------------------------------------------------------------------------
// Kernel 3: QKV projection GEMM.  C[4096x1024] = X[4096x1024] @ W + bias.
// X bf16 row-major, Wt bf16 as Wt[n][k].  Block = 128 thr = 4 waves; each
// workgroup does a 64x64 tile; wave w owns the n-strip [w*16, w*16+16).
// vmode==0: store blocked (b,h,nb)[i][d]; vmode==1: store transposed [d][i].
// ---------------------------------------------------------------------------
__global__ __launch_bounds__(128) void gemm_qkv_kernel(
    const unsigned short* __restrict__ X, const unsigned short* __restrict__ Wt,
    const float* __restrict__ bias, unsigned short* __restrict__ dst, int vmode) {
  int mb = blockIdx.x;              // 64 m-tiles of 64 rows
  int nb = blockIdx.y;              // 16 n-tiles of 64 cols
  int tid = threadIdx.x, lane = tid & 31, wave = tid >> 5;
  int hi = lane >> 4;               // half-wave select
  int l15 = lane & 15;
  int nlane = nb * 64 + wave * 16 + l15;

  v8f acc[4] = {};
  const unsigned short* wrow = Wt + (size_t)nlane * HID;
  for (int k0 = 0; k0 < HID; k0 += 32) {
    BF16Frag bfrag;
    const unsigned short* bp = wrow + k0 + hi * 8;      // two 16B chunks, 32 bytes apart
    bfrag.q[0] = *(const u32x4*)bp;
    bfrag.q[1] = *(const u32x4*)(bp + 16);
#pragma unroll
    for (int t = 0; t < 4; ++t) {
      BF16Frag a;
      const unsigned short* ap =
          X + (size_t)(mb * 64 + t * 16 + l15) * HID + k0 + hi * 8;
      a.q[0] = *(const u32x4*)ap;
      a.q[1] = *(const u32x4*)(ap + 16);
      acc[t] = wmma_bf16(a, bfrag, acc[t]);
    }
  }
  float bn = bias[nlane];
  int hcol = nlane >> 6, d = nlane & 63;
#pragma unroll
  for (int t = 0; t < 4; ++t) {
#pragma unroll
    for (int r = 0; r < 8; ++r) {
      int row = mb * 64 + t * 16 + hi * 8 + r;     // 0..4095 = b*S + s
      int b = row >> 11;
      int s = row & 2047;
      int nbq = s >> 6, i = s & 63;
      size_t base = (((size_t)b * HH + hcol) * NBQ + nbq) * (BLK * DD);
      size_t off = vmode ? (base + (size_t)d * BLK + i)
                         : (base + (size_t)i * DD + d);
      dst[off] = f2bf(acc[t][r] + bn);
    }
  }
}

// ---------------------------------------------------------------------------
// Kernel 4: block-sparse attention.  One workgroup per (b,h,qblock).
// 256 threads = 8 waves.  Wave w: q-subtile qi = w>>1, j-half jh = w&1.
// Gather phase uses GLOBAL_LOAD_ASYNC_TO_LDS_B128 (ASYNCcnt-tracked).
// ---------------------------------------------------------------------------
__global__ __launch_bounds__(256) void attn_kernel(
    const unsigned short* __restrict__ Qb, const unsigned short* __restrict__ Kb,
    const unsigned short* __restrict__ Vtb, const float* __restrict__ mask,
    const int* __restrict__ kidx, float* __restrict__ out) {
  __shared__ __align__(16) unsigned short Qs[BLK * DD];        //  8 KB  [i][d]
  __shared__ __align__(16) unsigned short Ks[SLEN * DD];       // 64 KB  [jg][d] -> reused as P [i][jg]
  __shared__ __align__(16) unsigned short Vts[DD * SLEN];      // 64 KB  [d][jg]
  __shared__ float rmaxred[BLK][2];
  __shared__ float rsumred[BLK][2];
  __shared__ int kbs[KK];

  int wg = blockIdx.x;                       // (b*H + h)*NBQ + qb
  int qb = wg % NBQ;
  int bh = wg / NBQ;
  int h = bh % HH;
  int tid = threadIdx.x, lane = tid & 31, wave = tid >> 5;
  int hi = lane >> 4, l15 = lane & 15;

  if (tid < KK) kbs[tid] = kidx[(h * NBQ + qb) * KK + tid];
  // Q block: 4096 ushorts = 512 x 16B chunks, async direct to LDS
  const unsigned short* qsrc = Qb + (size_t)wg * (BLK * DD);
  for (int i = tid; i < 512; i += 256)
    async_copy_b128(lds_off(Qs + i * 8), qsrc + i * 8);
  wait_asynccnt0();
  __syncthreads();   // kbs visible to all

  // Gather K and V^T blocks into LDS (async, 16B per lane per chunk)
  for (int k = 0; k < KK; ++k) {
    int kb = kbs[k];
    const unsigned short* ksrc = Kb + ((size_t)bh * NBQ + kb) * (BLK * DD);
    for (int i = tid; i < 512; i += 256)
      async_copy_b128(lds_off(Ks + k * (BLK * DD) + i * 8), ksrc + i * 8);
    const unsigned short* vsrc = Vtb + ((size_t)bh * NBQ + kb) * (BLK * DD);
    for (int i = tid; i < 512; i += 256) {
      int d = i >> 3, c = i & 7;      // source row d (64 ushorts = 8 chunks)
      async_copy_b128(lds_off(Vts + d * SLEN + k * BLK + c * 8), vsrc + i * 8);
    }
  }
  wait_asynccnt0();
  __syncthreads();

  // ---- phase 1: S = Q K^T for this wave's 16x256 strip, kept in VGPRs ----
  int qi = wave >> 1;      // 0..3
  int jh = wave & 1;       // 0..1  (columns jh*256 .. +255)
  int arow = qi * 16 + l15;
  v8f acc[16] = {};
#pragma unroll
  for (int ji = 0; ji < 16; ++ji) {
    int jt = jh * 16 + ji;                 // global j-tile 0..31
    int brow = jt * 16 + l15;              // gathered key index 0..511
#pragma unroll
    for (int d0 = 0; d0 < DD; d0 += 32) {
      BF16Frag a, b;
      const unsigned short* ap = Qs + arow * DD + d0 + hi * 8;
      a.q[0] = *(const u32x4*)ap;
      a.q[1] = *(const u32x4*)(ap + 16);
      const unsigned short* bp = Ks + brow * DD + d0 + hi * 8;
      b.q[0] = *(const u32x4*)bp;
      b.q[1] = *(const u32x4*)(bp + 16);
      acc[ji] = wmma_bf16(a, b, acc[ji]);
    }
  }

  // ---- phase 2: scale * mask (multiplicative, per reference) ----
  const float scale = 0.125f;              // 1/sqrt(D)
  int mrow0 = qb * 64 + qi * 16 + hi * 8;  // global query row base
#pragma unroll
  for (int ji = 0; ji < 16; ++ji) {
    int jt = jh * 16 + ji;
    int kb = kbs[jt >> 2];
    int mcol = kb * 64 + (jt & 3) * 16 + l15;
#pragma unroll
    for (int r = 0; r < 8; ++r) {
      float m = mask[(size_t)(mrow0 + r) * SS + mcol];
      acc[ji][r] = acc[ji][r] * scale * m;
    }
  }

  // ---- phase 3: softmax over the 512 gathered keys ----
#pragma unroll
  for (int r = 0; r < 8; ++r) {
    float m = acc[0][r];
#pragma unroll
    for (int ji = 1; ji < 16; ++ji) m = fmaxf(m, acc[ji][r]);
#pragma unroll
    for (int off = 1; off < 16; off <<= 1)
      m = fmaxf(m, __shfl_xor(m, off, 32));
    if (l15 == 0) rmaxred[qi * 16 + hi * 8 + r][jh] = m;
  }
  __syncthreads();
#pragma unroll
  for (int r = 0; r < 8; ++r) {
    int i = qi * 16 + hi * 8 + r;
    float rowmax = fmaxf(rmaxred[i][0], rmaxred[i][1]);
    float s = 0.f;
#pragma unroll
    for (int ji = 0; ji < 16; ++ji) {
      float e = __expf(acc[ji][r] - rowmax);
      acc[ji][r] = e;                      // keep exp() in registers
      s += e;
    }
#pragma unroll
    for (int off = 1; off < 16; off <<= 1)
      s += __shfl_xor(s, off, 32);
    if (l15 == 0) rsumred[i][jh] = s;
  }
  __syncthreads();

  // ---- phase 4: normalize, write P (bf16) into LDS reusing Ks buffer ----
  unsigned short* Pb = Ks;                 // [i][jg], stride SLEN
#pragma unroll
  for (int r = 0; r < 8; ++r) {
    int i = qi * 16 + hi * 8 + r;
    float inv = 1.0f / (rsumred[i][0] + rsumred[i][1]);
#pragma unroll
    for (int ji = 0; ji < 16; ++ji) {
      int col = jh * 256 + ji * 16 + l15;
      Pb[i * SLEN + col] = f2bf(acc[ji][r] * inv);
    }
  }
  __syncthreads();

  // ---- phase 5: ctx = P @ Vg  (P 64x512 bf16, V^T tiles in Vts) ----
  int b = bh / HH;
  for (int t = wave; t < 16; t += 8) {     // 4 q-tiles x 4 d-tiles
    int tqi = t >> 2, di = t & 3;
    int prow = tqi * 16 + l15;
    int vrow = di * 16 + l15;
    v8f cacc = {};
#pragma unroll
    for (int j0 = 0; j0 < SLEN; j0 += 32) {
      BF16Frag a, bb;
      const unsigned short* ap = Pb + prow * SLEN + j0 + hi * 8;
      a.q[0] = *(const u32x4*)ap;
      a.q[1] = *(const u32x4*)(ap + 16);
      const unsigned short* bp = Vts + vrow * SLEN + j0 + hi * 8;
      bb.q[0] = *(const u32x4*)bp;
      bb.q[1] = *(const u32x4*)(bp + 16);
      cacc = wmma_bf16(a, bb, cacc);
    }
    int dcol = di * 16 + l15;
#pragma unroll
    for (int r = 0; r < 8; ++r) {
      int i = tqi * 16 + hi * 8 + r;
      out[((size_t)b * SS + qb * 64 + i) * HID + h * 64 + dcol] = cacc[r];
    }
  }
}

// ---------------------------------------------------------------------------
// Host launcher
// ---------------------------------------------------------------------------
extern "C" void kernel_launch(void* const* d_in, const int* in_sizes, int n_in,
                              void* d_out, int out_size, void* d_ws, size_t ws_size,
                              hipStream_t stream) {
  const float* hidden = (const float*)d_in[0];
  const float* mask   = (const float*)d_in[1];
  const int*   kidx   = (const int*)d_in[2];
  const float* Wq = (const float*)d_in[3];
  const float* bq = (const float*)d_in[4];
  const float* Wk = (const float*)d_in[5];
  const float* bk = (const float*)d_in[6];
  const float* Wv = (const float*)d_in[7];
  const float* bv = (const float*)d_in[8];
  float* out = (float*)d_out;
  (void)in_sizes; (void)n_in; (void)out_size; (void)ws_size;

  char* ws = (char*)d_ws;
  // workspace layout (bytes):
  //   [0,  8M)  X bf16          (4096 x 1024)
  //   [8M, 14M) Wt q/k/v bf16   (3 x 1024 x 1024)
  //   [14M,22M) Q  bf16 blocked (b,h,nb)[i][d]
  //   [22M,30M) K  bf16 blocked (b,h,nb)[i][d]
  //   [30M,38M) V^T bf16 blocked (b,h,nb)[d][i]
  unsigned short* X   = (unsigned short*)(ws);
  unsigned short* Wtq = (unsigned short*)(ws + ((size_t)8  << 20));
  unsigned short* Wtk = Wtq + (size_t)HID * HID;
  unsigned short* Wtv = Wtk + (size_t)HID * HID;
  unsigned short* Qb  = (unsigned short*)(ws + ((size_t)14 << 20));
  unsigned short* Kb  = (unsigned short*)(ws + ((size_t)22 << 20));
  unsigned short* Vtb = (unsigned short*)(ws + ((size_t)30 << 20));

  const int nX = BB * SS * HID;   // 4,194,304
  cvt_f32_bf16_kernel<<<4096, 256, 0, stream>>>(hidden, X, nX);

  dim3 tgrid(16, 16);
  transpose_w_kernel<<<tgrid, 256, 0, stream>>>(Wq, Wtq);
  transpose_w_kernel<<<tgrid, 256, 0, stream>>>(Wk, Wtk);
  transpose_w_kernel<<<tgrid, 256, 0, stream>>>(Wv, Wtv);

  dim3 ggrid(64, 16);
  gemm_qkv_kernel<<<ggrid, 128, 0, stream>>>(X, Wtq, bq, Qb, 0);
  gemm_qkv_kernel<<<ggrid, 128, 0, stream>>>(X, Wtk, bk, Kb, 0);
  gemm_qkv_kernel<<<ggrid, 128, 0, stream>>>(X, Wtv, bv, Vtb, 1);

  attn_kernel<<<BB * HH * NBQ, 256, 0, stream>>>(Qb, Kb, Vtb, mask, kidx, out);
}
